// CloAttention_47304769798497
// MI455X (gfx1250) — matrix operand order repaired and checked by reading, not compiled, not run on hardware
//
#include <hip/hip_runtime.h>
#include <math.h>

// ---------------------------------------------------------------------------
// CloFormer attention for MI455X (gfx1250): all channel-mixing matmuls via
// v_wmma_f32_16x16x32_bf16 (wave32); GEMM tiles staged into LDS with the
// CDNA5 async-copy path (GLOBAL_LOAD_ASYNC_TO_LDS_B128 + ASYNCcnt) when the
// toolchain exposes it, double-buffered; b128-vectorized fragment loads;
// LDS-staged coalesced NCHW writeback for the projection output.
// ---------------------------------------------------------------------------

typedef __attribute__((ext_vector_type(16))) __bf16 v16bf;
typedef __attribute__((ext_vector_type(8)))  float  v8f;

#define HWPIX 3136          // 56*56
#define WIMG  56
#define DIMC  256
#define BATCH 32
#define NPIX  (BATCH * HWPIX)   // 100352
#define SCALER 0.17677669529663687f   // 32^-0.5

#ifndef __has_builtin
#define __has_builtin(x) 0
#endif
#if __has_builtin(__builtin_amdgcn_global_load_async_to_lds_b128)
#define HAVE_ASYNC_LDS 1
#else
#define HAVE_ASYNC_LDS 0
#endif

typedef int v4i_ __attribute__((vector_size(16)));
typedef __attribute__((address_space(3))) v4i_ lds_v4i;

__device__ __forceinline__ void async_b128(const void* g, void* l) {
#if HAVE_ASYNC_LDS
    __builtin_amdgcn_global_load_async_to_lds_b128((v4i_*)g, (lds_v4i*)l, 0, 0);
#endif
}
__device__ __forceinline__ void wait_async0() {
#if HAVE_ASYNC_LDS
#if __has_builtin(__builtin_amdgcn_s_wait_asynccnt)
    __builtin_amdgcn_s_wait_asynccnt(0);
#else
    asm volatile("s_wait_asynccnt 0" ::: "memory");
#endif
#endif
}

__device__ __forceinline__ unsigned short f2bf(float f) {
    unsigned int u = __float_as_uint(f);
    unsigned int r = u + 0x7FFFu + ((u >> 16) & 1u);   // round-nearest-even
    return (unsigned short)(r >> 16);
}
__device__ __forceinline__ float bf2f(unsigned short h) {
    return __uint_as_float(((unsigned int)h) << 16);
}

union FragV {           // WMMA operand as two 16B vectors
    v16bf v;
    uint4 q[2];
    unsigned short u[16];
};

// ---------------------------------------------------------------------------
// Weight convert (bf16); weights stay (N rows, K cols) = n-major/k-contiguous,
// which is exactly the B-fragment-friendly layout.
// ---------------------------------------------------------------------------
__global__ void convert_w(const float* __restrict__ w,
                          unsigned short* __restrict__ w16, int n) {
    int idx = blockIdx.x * 256 + threadIdx.x;
    if (idx < n) w16[idx] = f2bf(w[idx]);
}

// x (B,C,H,W) f32 -> x16 [b*HW+q][c] bf16; 32x32 LDS tile, both sides coalesced
__global__ void pack_x(const float* __restrict__ x, unsigned short* __restrict__ x16) {
    __shared__ float tile[32][33];
    const int t = threadIdx.x;
    const int lq = t & 31, r8 = t >> 5;               // 0..31, 0..7
    const int q0 = blockIdx.x * 32, c0 = blockIdx.y * 32, b = blockIdx.z;
    #pragma unroll
    for (int it = 0; it < 4; it++) {
        int c = c0 + r8 + it * 8;
        tile[r8 + it * 8][lq] = x[((size_t)(b * DIMC + c)) * HWPIX + q0 + lq];
    }
    __syncthreads();
    #pragma unroll
    for (int it = 0; it < 4; it++) {
        int q = q0 + r8 + it * 8;
        x16[((size_t)b * HWPIX + q) * DIMC + c0 + lq] = f2bf(tile[lq][r8 + it * 8]);
    }
}

// 7x7 average pool: x (B,C,56,56) f32 -> pooled [b*64+p][c] bf16
__global__ void pool_x(const float* __restrict__ x, unsigned short* __restrict__ pooled) {
    int idx = blockIdx.x * 256 + threadIdx.x;   // 2048*256
    int c = idx & 255;
    int bp = idx >> 8;
    int b = bp >> 6, p = bp & 63;
    int ph = p >> 3, pw = p & 7;
    const float* src = x + ((size_t)(b * DIMC + c)) * HWPIX + (ph * 7) * WIMG + pw * 7;
    float s = 0.f;
    #pragma unroll
    for (int r = 0; r < 7; r++)
        #pragma unroll
        for (int cc = 0; cc < 7; cc++)
            s += src[r * WIMG + cc];
    pooled[(size_t)bp * DIMC + c] = f2bf(s * (1.f / 49.f));
}

// ---------------------------------------------------------------------------
// Depthwise 3x3 conv on qkv-linear (pixel-major bf16, 384ch) + bias,
// emits qk = q*k*SCALER (bf16) and v (bf16), both pixel-major 128ch.
// (qkv buffer is 77MB -> the 9x re-read hits the 192MB L2.)
// ---------------------------------------------------------------------------
__global__ void dwconv_qkv(const unsigned short* __restrict__ qkv,
                           const float* __restrict__ dww,
                           const float* __restrict__ dwb,
                           unsigned short* __restrict__ qk,
                           unsigned short* __restrict__ vout) {
    int idx = blockIdx.x * 256 + threadIdx.x;   // n*128 + c
    int c = idx & 127;
    int n = idx >> 7;
    if (n >= NPIX) return;
    int b = n / HWPIX, q = n - b * HWPIX;
    int h = q / WIMG, w = q - h * WIMG;
    float aq = 0.f, ak = 0.f, av = 0.f;
    #pragma unroll
    for (int kh = 0; kh < 3; kh++) {
        int hh = h + kh - 1;
        if (hh < 0 || hh >= WIMG) continue;
        #pragma unroll
        for (int kw = 0; kw < 3; kw++) {
            int ww = w + kw - 1;
            if (ww < 0 || ww >= WIMG) continue;
            size_t base = ((size_t)(b * HWPIX + hh * WIMG + ww)) * 384;
            int wi = kh * 3 + kw;
            aq += bf2f(qkv[base + c])        * dww[(size_t)(c)       * 9 + wi];
            ak += bf2f(qkv[base + 128 + c])  * dww[(size_t)(c + 128) * 9 + wi];
            av += bf2f(qkv[base + 256 + c])  * dww[(size_t)(c + 256) * 9 + wi];
        }
    }
    aq += dwb[c]; ak += dwb[c + 128]; av += dwb[c + 256];
    qk[(size_t)n * 128 + c]   = f2bf(aq * ak * SCALER);
    vout[(size_t)n * 128 + c] = f2bf(av);
}

// ---------------------------------------------------------------------------
// Tiled WMMA GEMM: C[M,N] = A[M,K](bf16 row-major) * W[N,K](bf16 n-major)
// Block tile 64x64, 4 waves (128 thr), K-step 32, double-buffered async LDS.
// M%64==0, N%64==0, K%32==0 guaranteed by the workload.
// ---------------------------------------------------------------------------
enum { EPI_NONE = 0, EPI_HSW = 1, EPI_TANHV = 2, EPI_SCATTER = 3 };

template <int EPI>
__global__ __launch_bounds__(128)
void gemm_bf16(const unsigned short* __restrict__ A,
               const unsigned short* __restrict__ Wn,   // (N,K) n-major
               int M, int N, int K,
               const float* __restrict__ bias,
               const unsigned short* __restrict__ vbuf,
               unsigned short* __restrict__ outb,
               float* __restrict__ outf,
               int outStride, int outColOff) {
    __shared__ __align__(16) unsigned short As[2][64][40];   // [buf][m][k]
    __shared__ __align__(16) unsigned short Bs[2][64][40];   // [buf][n][k]
    __shared__ __align__(16) float Ts[(EPI == EPI_SCATTER) ? 64 : 1][68];

    const int t = threadIdx.x;
    const int lane = t & 31, wave = t >> 5;
    const int hf = lane >> 4, ln = lane & 15;
    const int row0 = blockIdx.x * 64, col0 = blockIdx.y * 64;

    v8f acc[4];
    #pragma unroll
    for (int i = 0; i < 4; i++)
        #pragma unroll
        for (int j = 0; j < 8; j++) acc[i][j] = 0.f;

    const int cr = t >> 1, ch = t & 1;   // 64 rows x 2 half-rows of 16

#if HAVE_ASYNC_LDS
    auto issueTile = [&](int buf, int k0) {
        const unsigned short* ga = A  + (size_t)(row0 + cr) * K + k0 + ch * 16;
        const unsigned short* gb = Wn + (size_t)(col0 + cr) * K + k0 + ch * 16;
        async_b128(ga,     &As[buf][cr][ch * 16]);
        async_b128(ga + 8, &As[buf][cr][ch * 16 + 8]);
        async_b128(gb,     &Bs[buf][cr][ch * 16]);
        async_b128(gb + 8, &Bs[buf][cr][ch * 16 + 8]);
    };
    issueTile(0, 0);
    int buf = 0;
    for (int k0 = 0; k0 < K; k0 += 32) {
        wait_async0();          // this wave's pending tile is in LDS
        __syncthreads();        // all waves' pieces visible
        if (k0 + 32 < K) issueTile(buf ^ 1, k0 + 32);

        FragV a;
        a.q[0] = *(const uint4*)&As[buf][wave * 16 + ln][hf * 8];
        a.q[1] = *(const uint4*)&As[buf][wave * 16 + ln][hf * 8 + 16];
        #pragma unroll
        for (int nt = 0; nt < 4; nt++) {
            FragV b;
            b.q[0] = *(const uint4*)&Bs[buf][nt * 16 + ln][hf * 8];
            b.q[1] = *(const uint4*)&Bs[buf][nt * 16 + ln][hf * 8 + 16];
            acc[nt] = __builtin_amdgcn_wmma_f32_16x16x32_bf16(
                false, a.v, false, b.v, (short)0, acc[nt], false, false);
        }
        __syncthreads();        // done reading buf before it is overwritten
        buf ^= 1;
    }
#else
    for (int k0 = 0; k0 < K; k0 += 32) {
        const uint4* ga = (const uint4*)(A  + (size_t)(row0 + cr) * K + k0 + ch * 16);
        uint4* la = (uint4*)&As[0][cr][ch * 16];
        la[0] = ga[0]; la[1] = ga[1];
        const uint4* gb = (const uint4*)(Wn + (size_t)(col0 + cr) * K + k0 + ch * 16);
        uint4* lb = (uint4*)&Bs[0][cr][ch * 16];
        lb[0] = gb[0]; lb[1] = gb[1];
        if (k0 + 32 < K) {   // hint the next K tile (global_prefetch_b8)
            __builtin_prefetch(A  + (size_t)(row0 + cr) * K + k0 + 32 + ch * 16, 0, 1);
            __builtin_prefetch(Wn + (size_t)(col0 + cr) * K + k0 + 32 + ch * 16, 0, 1);
        }
        __syncthreads();
        FragV a;
        a.q[0] = *(const uint4*)&As[0][wave * 16 + ln][hf * 8];
        a.q[1] = *(const uint4*)&As[0][wave * 16 + ln][hf * 8 + 16];
        #pragma unroll
        for (int nt = 0; nt < 4; nt++) {
            FragV b;
            b.q[0] = *(const uint4*)&Bs[0][nt * 16 + ln][hf * 8];
            b.q[1] = *(const uint4*)&Bs[0][nt * 16 + ln][hf * 8 + 16];
            acc[nt] = __builtin_amdgcn_wmma_f32_16x16x32_bf16(
                false, a.v, false, b.v, (short)0, acc[nt], false, false);
        }
        __syncthreads();
    }
#endif

    if (EPI == EPI_SCATTER) {
        // Stage 64x64 f32 tile (channel-major) in LDS, then write NCHW rows
        // with 128B-contiguous stores per thread.
        #pragma unroll
        for (int nt = 0; nt < 4; nt++)
            #pragma unroll
            for (int i = 0; i < 8; i++)
                Ts[nt * 16 + ln][wave * 16 + hf * 8 + i] = acc[nt][i];
        __syncthreads();
        const int c2 = t >> 1, h2 = t & 1;
        const int bimg = row0 / HWPIX;
        const int qb = row0 - bimg * HWPIX;   // tile never crosses an image
        float* dst = outf + ((size_t)bimg * DIMC + col0 + c2) * HWPIX + qb + h2 * 32;
        #pragma unroll
        for (int j = 0; j < 32; j += 4)
            *(float4*)(dst + j) = *(const float4*)&Ts[c2][h2 * 32 + j];
    } else {
        // C layout: row = hf*8 + i, col = ln, per 16-col tile.
        #pragma unroll
        for (int nt = 0; nt < 4; nt++) {
            #pragma unroll
            for (int i = 0; i < 8; i++) {
                int gr = row0 + wave * 16 + hf * 8 + i;
                int gc = col0 + nt * 16 + ln;
                float v = acc[nt][i];
                if (EPI == EPI_NONE) {
                    outb[(size_t)gr * outStride + outColOff + gc] = f2bf(v);
                } else if (EPI == EPI_HSW) {
                    v += bias[gc];
                    float c = fminf(fmaxf(v + 3.f, 0.f), 6.f);
                    outb[(size_t)gr * outStride + gc] = f2bf(v * c * (1.f / 6.f));
                } else {   // EPI_TANHV
                    v += bias[gc];
                    v = tanhf(v) * bf2f(vbuf[(size_t)gr * 128 + gc]);
                    outb[(size_t)gr * outStride + outColOff + gc] = f2bf(v);
                }
            }
        }
    }
}

// ---------------------------------------------------------------------------
// Fused global attention per (b, head): scores = gq @ gk^T * s (WMMA),
// softmax over 64 pooled keys (LDS), out = P @ gv (WMMA) -> cat[:,128:256].
// Block: 128 threads = 4 waves; each wave owns 16 query rows; grid.x tiles 64.
// ---------------------------------------------------------------------------
__global__ __launch_bounds__(128)
void attn_global(const unsigned short* __restrict__ gq,
                 const unsigned short* __restrict__ kv,
                 unsigned short* __restrict__ cat) {
    __shared__ __align__(16) unsigned short tK[64][40];    // [p][d] (+pad)
    __shared__ __align__(16) unsigned short tVt[32][72];   // [d][p] (+pad)
    __shared__ float          sc[4][16][64];
    __shared__ __align__(16) unsigned short pA[4][16][64];

    const int t = threadIdx.x, lane = t & 31, wave = t >> 5;
    const int hf = lane >> 4, ln = lane & 15;
    const int bl = blockIdx.y;            // b*4 + l
    const int b = bl >> 2, l = bl & 3;
    const int q0 = blockIdx.x * 64;

    {   // K rows copied directly (b128); tiny V tile transposed via LDS scatter
        const int p = t >> 1, hh = t & 1;
        const size_t kbase = ((size_t)(b * 64 + p)) * 256 + l * 32 + hh * 16;
        const uint4* gk = (const uint4*)(kv + kbase);
        uint4* dk = (uint4*)&tK[p][hh * 16];
        dk[0] = gk[0]; dk[1] = gk[1];
        FragV tv;
        tv.q[0] = *(const uint4*)(kv + kbase + 128);
        tv.q[1] = *(const uint4*)(kv + kbase + 128 + 8);
        #pragma unroll
        for (int j = 0; j < 16; j++) tVt[hh * 16 + j][p] = tv.u[j];
    }
    __syncthreads();

    const size_t rowbase = ((size_t)b * HWPIX + q0 + wave * 16 + ln) * 128 + l * 32;
    FragV a;
    a.q[0] = *(const uint4*)(gq + rowbase + hf * 8);
    a.q[1] = *(const uint4*)(gq + rowbase + hf * 8 + 16);

    v8f s[4];
    #pragma unroll
    for (int nt = 0; nt < 4; nt++)
        #pragma unroll
        for (int i = 0; i < 8; i++) s[nt][i] = 0.f;

    #pragma unroll
    for (int nt = 0; nt < 4; nt++) {
        FragV bk;   // B[k=d][n=p]: column p = nt*16+ln, d contiguous
        bk.q[0] = *(const uint4*)&tK[nt * 16 + ln][hf * 8];
        bk.q[1] = *(const uint4*)&tK[nt * 16 + ln][hf * 8 + 16];
        s[nt] = __builtin_amdgcn_wmma_f32_16x16x32_bf16(
            false, a.v, false, bk.v, (short)0, s[nt], false, false);
    }
    #pragma unroll
    for (int nt = 0; nt < 4; nt++)
        #pragma unroll
        for (int i = 0; i < 8; i++)
            sc[wave][hf * 8 + i][nt * 16 + ln] = s[nt][i] * SCALER;
    __syncthreads();

    if (lane < 16) {   // one lane per query row: softmax over 64 keys
        float* r = sc[wave][lane];
        float m = r[0];
        for (int j = 1; j < 64; j++) m = fmaxf(m, r[j]);
        float sum = 0.f;
        for (int j = 0; j < 64; j++) sum += __expf(r[j] - m);
        float inv = 1.f / sum;
        for (int j = 0; j < 64; j++) pA[wave][lane][j] = f2bf(__expf(r[j] - m) * inv);
    }
    __syncthreads();

    v8f o[2];
    #pragma unroll
    for (int nt = 0; nt < 2; nt++)
        #pragma unroll
        for (int i = 0; i < 8; i++) o[nt][i] = 0.f;

    #pragma unroll
    for (int kk = 0; kk < 64; kk += 32) {
        FragV pa;
        pa.q[0] = *(const uint4*)&pA[wave][ln][kk + hf * 8];
        pa.q[1] = *(const uint4*)&pA[wave][ln][kk + hf * 8 + 16];
        #pragma unroll
        for (int nt = 0; nt < 2; nt++) {
            FragV bv;   // B[k=p][n=d]: column d = nt*16+ln, p contiguous in tVt
            bv.q[0] = *(const uint4*)&tVt[nt * 16 + ln][kk + hf * 8];
            bv.q[1] = *(const uint4*)&tVt[nt * 16 + ln][kk + hf * 8 + 16];
            o[nt] = __builtin_amdgcn_wmma_f32_16x16x32_bf16(
                false, pa.v, false, bv.v, (short)0, o[nt], false, false);
        }
    }

    #pragma unroll
    for (int nt = 0; nt < 2; nt++)
        #pragma unroll
        for (int i = 0; i < 8; i++) {
            int pix = b * HWPIX + q0 + wave * 16 + hf * 8 + i;
            int d = nt * 16 + ln;
            cat[(size_t)pix * 256 + 128 + l * 32 + d] = f2bf(o[nt][i]);
        }
}

// ---------------------------------------------------------------------------
extern "C" void kernel_launch(void* const* d_in, const int* in_sizes, int n_in,
                              void* d_out, int out_size, void* d_ws, size_t ws_size,
                              hipStream_t stream) {
    const float* x      = (const float*)d_in[0];
    const float* qkv_w  = (const float*)d_in[1];
    const float* dw_w   = (const float*)d_in[2];
    const float* dw_b   = (const float*)d_in[3];
    const float* act1_w = (const float*)d_in[4];
    const float* act1_b = (const float*)d_in[5];
    const float* act2_w = (const float*)d_in[6];
    const float* act2_b = (const float*)d_in[7];
    const float* gq_w   = (const float*)d_in[8];
    const float* gkv_w  = (const float*)d_in[9];
    const float* proj_w = (const float*)d_in[10];
    float* out = (float*)d_out;
    char* ws = (char*)d_ws;

    size_t off = 0;
    auto alloc = [&](size_t bytes) {
        size_t o = off; off += (bytes + 255) & ~(size_t)255; return o;
    };
    const size_t SZ_M128 = (size_t)NPIX * 128 * 2;
    unsigned short* X16  = (unsigned short*)(ws + alloc((size_t)NPIX * 256 * 2));
    size_t offQKV        = alloc((size_t)NPIX * 384 * 2);     // reused below
    unsigned short* QKV  = (unsigned short*)(ws + offQKV);
    unsigned short* A1   = (unsigned short*)(ws + offQKV);            // reuse
    unsigned short* GQ   = (unsigned short*)(ws + offQKV + SZ_M128);  // reuse
    unsigned short* QK   = (unsigned short*)(ws + alloc(SZ_M128));
    unsigned short* VB   = (unsigned short*)(ws + alloc(SZ_M128));
    unsigned short* CAT  = (unsigned short*)(ws + alloc((size_t)NPIX * 256 * 2));
    unsigned short* POOL = (unsigned short*)(ws + alloc((size_t)2048 * 256 * 2));
    unsigned short* KV   = (unsigned short*)(ws + alloc((size_t)2048 * 256 * 2));
    unsigned short* Wqkv = (unsigned short*)(ws + alloc((size_t)384 * 256 * 2));
    unsigned short* Wa1  = (unsigned short*)(ws + alloc((size_t)128 * 128 * 2));
    unsigned short* Wa2  = (unsigned short*)(ws + alloc((size_t)128 * 128 * 2));
    unsigned short* Wgq  = (unsigned short*)(ws + alloc((size_t)128 * 256 * 2));
    unsigned short* Wgkv = (unsigned short*)(ws + alloc((size_t)256 * 256 * 2));
    unsigned short* Wprj = (unsigned short*)(ws + alloc((size_t)256 * 256 * 2));
    (void)ws_size; (void)in_sizes; (void)n_in; (void)out_size;

    convert_w<<<(384 * 256 + 255) / 256, 256, 0, stream>>>(qkv_w,  Wqkv, 384 * 256);
    convert_w<<<(128 * 128 + 255) / 256, 256, 0, stream>>>(act1_w, Wa1,  128 * 128);
    convert_w<<<(128 * 128 + 255) / 256, 256, 0, stream>>>(act2_w, Wa2,  128 * 128);
    convert_w<<<(128 * 256 + 255) / 256, 256, 0, stream>>>(gq_w,   Wgq,  128 * 256);
    convert_w<<<(256 * 256 + 255) / 256, 256, 0, stream>>>(gkv_w,  Wgkv, 256 * 256);
    convert_w<<<(256 * 256 + 255) / 256, 256, 0, stream>>>(proj_w, Wprj, 256 * 256);

    pack_x<<<dim3(HWPIX / 32, DIMC / 32, BATCH), 256, 0, stream>>>(x, X16);

    gemm_bf16<EPI_NONE><<<dim3(NPIX / 64, 384 / 64), 128, 0, stream>>>(
        X16, Wqkv, NPIX, 384, 256, nullptr, nullptr, QKV, nullptr, 384, 0);

    dwconv_qkv<<<(unsigned)(((size_t)NPIX * 128) / 256), 256, 0, stream>>>(
        QKV, dw_w, dw_b, QK, VB);

    gemm_bf16<EPI_HSW><<<dim3(NPIX / 64, 128 / 64), 128, 0, stream>>>(
        QK, Wa1, NPIX, 128, 128, act1_b, nullptr, A1, nullptr, 128, 0);

    gemm_bf16<EPI_TANHV><<<dim3(NPIX / 64, 128 / 64), 128, 0, stream>>>(
        A1, Wa2, NPIX, 128, 128, act2_b, VB, CAT, nullptr, 256, 0);

    gemm_bf16<EPI_NONE><<<dim3(NPIX / 64, 128 / 64), 128, 0, stream>>>(
        X16, Wgq, NPIX, 128, 256, nullptr, nullptr, GQ, nullptr, 128, 0);

    pool_x<<<(2048 * 256) / 256, 256, 0, stream>>>(x, POOL);

    gemm_bf16<EPI_NONE><<<dim3(2048 / 64, 256 / 64), 128, 0, stream>>>(
        POOL, Wgkv, 2048, 256, 256, nullptr, nullptr, KV, nullptr, 256, 0);

    attn_global<<<dim3(HWPIX / 64, BATCH * 4), 128, 0, stream>>>(GQ, KV, CAT);

    gemm_bf16<EPI_SCATTER><<<dim3(NPIX / 64, 256 / 64), 128, 0, stream>>>(
        CAT, Wprj, NPIX, 256, 256, nullptr, nullptr, nullptr, out, 0, 0);
}
